// UltraEncoderLayer_58935541235819
// MI455X (gfx1250) — compile-verified, matrix operand order checked by reading, not run on hardware
//
#include <hip/hip_runtime.h>
#include <math.h>

// Problem constants (from reference)
#define Bq   8
#define Lq   1024
#define Dq   512
#define Hq   8
#define HDq  64
#define DFFq 2048
#define Mrows (Bq * Lq)          // 8192

typedef __attribute__((ext_vector_type(16))) __bf16 v16bf;
typedef __attribute__((ext_vector_type(8)))  float  v8f;

__device__ __forceinline__ unsigned short f2bf(float f) {
  union { float f; unsigned u; } v; v.f = f;
  unsigned r = v.u + 0x7FFFu + ((v.u >> 16) & 1u);   // round-to-nearest-even
  return (unsigned short)(r >> 16);
}
__device__ __forceinline__ float bf2f(unsigned short h) {
  union { unsigned u; float f; } v; v.u = ((unsigned)h) << 16; return v.f;
}
__device__ __forceinline__ float sigmoidf(float x) { return 1.f / (1.f + __expf(-x)); }

union Frag { unsigned u[8]; v16bf v; };

__device__ __forceinline__ void load_bf16_frag(Frag& f, const unsigned short* p) {
  *(uint4*)(&f.u[0]) = *(const uint4*)p;
  *(uint4*)(&f.u[4]) = *(const uint4*)(p + 16);
}

// pack two f32 -> two bf16 (truncate) with a single v_perm_b32
__device__ __forceinline__ unsigned pk2(float lo, float hi) {
  return __builtin_amdgcn_perm(__float_as_uint(hi), __float_as_uint(lo), 0x07060302u);
}
__device__ __forceinline__ void load_f32_frag(Frag& f, const float* p) {
  float4 f0 = ((const float4*)p)[0];
  float4 f1 = ((const float4*)p)[1];
  float4 f2 = ((const float4*)(p + 16))[0];
  float4 f3 = ((const float4*)(p + 16))[1];
  f.u[0] = pk2(f0.x, f0.y); f.u[1] = pk2(f0.z, f0.w);
  f.u[2] = pk2(f1.x, f1.y); f.u[3] = pk2(f1.z, f1.w);
  f.u[4] = pk2(f2.x, f2.y); f.u[5] = pk2(f2.z, f2.w);
  f.u[6] = pk2(f3.x, f3.y); f.u[7] = pk2(f3.z, f3.w);
}

// ---------------------------------------------------------------------------
// bf16 WMMA GEMM, 32x32 C-tile per wave (2x2 grid of v_wmma_f32_16x16x32_bf16).
//   C[M,N] = A[M,K] @ Wt[N,K]^T * scale + bias
//   AF32:  A is f32 (converted to bf16 in-register via v_perm)
//   SPLIT: A = concat(A0, A1) along K at kSplit (gated-residual GEMMs)
//   Wt is pre-transposed weights bf16 [N,K].
//   Batched over blockIdx.z: A/B offset z*aBatch / z*bBatch;
//   C offset (z/Hdiv)*cOuter + (z%Hdiv)*cInner.
// ---------------------------------------------------------------------------
template <bool AF32, bool SPLIT>
__global__ void __launch_bounds__(256) wmma_gemm(
    const unsigned short* __restrict__ A0,
    const unsigned short* __restrict__ A1,
    const float*          __restrict__ Af,
    const unsigned short* __restrict__ Wt,
    const float*          __restrict__ bias,
    float*                __restrict__ Cf,
    unsigned short*       __restrict__ Cb,
    int M, int N, int K, int kSplit,
    int lda, int ldb, int ldc,
    long long aBatch, long long bBatch,
    long long cOuter, long long cInner, int Hdiv,
    const float* __restrict__ scale_ptr, float scale_const)
{
  const int lane  = threadIdx.x & 31;
  const int wave  = (int)(blockIdx.x * (blockDim.x >> 5) + (threadIdx.x >> 5));
  const int tilesN = N >> 5;
  const int tiles  = (M >> 5) * tilesN;
  if (wave >= tiles) return;
  const int tm = (wave / tilesN) << 5;
  const int tn = (wave % tilesN) << 5;
  const int z  = blockIdx.z;
  const long long aoff = (long long)z * aBatch;
  const long long boff = (long long)z * bBatch;
  const long long coff = (long long)(z / Hdiv) * cOuter + (long long)(z % Hdiv) * cInner;

  const int r0 = lane & 15;
  const int kh = (lane >> 4) << 3;     // 0 or 8
  const int m0 = tm + r0,  m1 = tm + 16 + r0;
  const int n0 = tn + r0,  n1 = tn + 16 + r0;

  v8f acc00 = {0.f,0.f,0.f,0.f,0.f,0.f,0.f,0.f};
  v8f acc01 = acc00, acc10 = acc00, acc11 = acc00;

  const unsigned short* pb0 = Wt + boff + (long long)n0 * ldb + kh;
  const unsigned short* pb1 = Wt + boff + (long long)n1 * ldb + kh;

  for (int k0 = 0; k0 < K; k0 += 32) {
    Frag a0, a1, b0, b1;
    load_bf16_frag(b0, pb0 + k0);
    load_bf16_frag(b1, pb1 + k0);
    const int kpf = (k0 + 32 < K) ? (k0 + 32) : k0;   // branchless clamp
    __builtin_prefetch(pb0 + kpf, 0, 1);
    __builtin_prefetch(pb1 + kpf, 0, 1);

    if (AF32) {
      const float* pa0 = Af + aoff + (long long)m0 * lda + k0 + kh;
      const float* pa1 = Af + aoff + (long long)m1 * lda + k0 + kh;
      load_f32_frag(a0, pa0);
      load_f32_frag(a1, pa1);
      __builtin_prefetch(Af + aoff + (long long)m0 * lda + kpf + kh, 0, 1);
      __builtin_prefetch(Af + aoff + (long long)m1 * lda + kpf + kh, 0, 1);
    } else {
      const unsigned short* base;
      int kk;
      if (SPLIT) {
        const bool lo = (k0 < kSplit);
        base = lo ? A0 : A1;
        kk   = lo ? k0 : (k0 - kSplit);
      } else { base = A0; kk = k0; }
      const unsigned short* pa0 = base + aoff + (long long)m0 * lda + kk + kh;
      const unsigned short* pa1 = base + aoff + (long long)m1 * lda + kk + kh;
      load_bf16_frag(a0, pa0);
      load_bf16_frag(a1, pa1);
      __builtin_prefetch(pa0 + 32, 0, 1);
      __builtin_prefetch(pa1 + 32, 0, 1);
    }

    acc00 = __builtin_amdgcn_wmma_f32_16x16x32_bf16(false, a0.v, false, b0.v, (short)0, acc00, false, false);
    acc01 = __builtin_amdgcn_wmma_f32_16x16x32_bf16(false, a0.v, false, b1.v, (short)0, acc01, false, false);
    acc10 = __builtin_amdgcn_wmma_f32_16x16x32_bf16(false, a1.v, false, b0.v, (short)0, acc10, false, false);
    acc11 = __builtin_amdgcn_wmma_f32_16x16x32_bf16(false, a1.v, false, b1.v, (short)0, acc11, false, false);
  }

  const float s   = scale_const * (scale_ptr ? scale_ptr[0] : 1.0f);
  const float bv0 = bias ? bias[n0] : 0.0f;
  const float bv1 = bias ? bias[n1] : 0.0f;
  const int   rr  = ((lane >> 4) << 3);

  #pragma unroll
  for (int r = 0; r < 8; ++r) {
    const int mmA = tm + r + rr, mmB = tm + 16 + r + rr;
    const long long cA0 = coff + (long long)mmA * ldc + n0;
    const long long cA1 = coff + (long long)mmA * ldc + n1;
    const long long cB0 = coff + (long long)mmB * ldc + n0;
    const long long cB1 = coff + (long long)mmB * ldc + n1;
    const float v00 = acc00[r] * s + bv0;
    const float v01 = acc01[r] * s + bv1;
    const float v10 = acc10[r] * s + bv0;
    const float v11 = acc11[r] * s + bv1;
    if (Cf) { Cf[cA0] = v00; Cf[cA1] = v01; Cf[cB0] = v10; Cf[cB1] = v11; }
    if (Cb) { Cb[cA0] = f2bf(v00); Cb[cA1] = f2bf(v01);
              Cb[cB0] = f2bf(v10); Cb[cB1] = f2bf(v11); }
  }
}

// ---------------------------------------------------------------------------
// Weight convert+transpose: W[K,N] f32 -> Wt[N,K] bf16
// ---------------------------------------------------------------------------
__global__ void transpose_to_bf16(const float* __restrict__ W,
                                  unsigned short* __restrict__ Wt, int K, int N)
{
  long long i = (long long)blockIdx.x * blockDim.x + threadIdx.x;
  if (i >= (long long)K * N) return;
  int n = (int)(i % N), k = (int)(i / N);
  Wt[(long long)n * K + k] = f2bf(W[i]);
}

__global__ void cvt_bf16(const float* __restrict__ in,
                         unsigned short* __restrict__ out, long long total)
{
  long long i = (long long)blockIdx.x * blockDim.x + threadIdx.x;
  if (i < total) out[i] = f2bf(in[i]);
}

// ---------------------------------------------------------------------------
// Positional MLP: pos_emb[L,D] = relu(pos @ w1 + b1) @ w2 + b2, pos = l/L
// ---------------------------------------------------------------------------
__global__ void __launch_bounds__(128) pos_emb_kernel(
    const float* __restrict__ w1, const float* __restrict__ b1,
    const float* __restrict__ w2, const float* __restrict__ b2,
    float* __restrict__ out)
{
  __shared__ float hid[128];
  const int l = blockIdx.x, tid = threadIdx.x;
  const float p = (float)l / (float)Lq;
  hid[tid] = fmaxf(p * w1[tid] + b1[tid], 0.f);
  __syncthreads();
  for (int t = 0; t < 4; ++t) {
    const int d = tid + (t << 7);
    float acc = b2[d];
    for (int j = 0; j < 128; ++j) acc += hid[j] * w2[j * Dq + d];
    out[(long long)l * Dq + d] = acc;
  }
}

// ---------------------------------------------------------------------------
// Rowwise LayerNorm (+optional positional add), bf16 output
// ---------------------------------------------------------------------------
__global__ void __launch_bounds__(256) ln_rows(
    const float* __restrict__ x, const float* __restrict__ w,
    const float* __restrict__ b, const float* __restrict__ extra,
    unsigned short* __restrict__ out)
{
  __shared__ float red[256];
  const int row = blockIdx.x, tid = threadIdx.x;
  const float* xr = x + (long long)row * Dq;
  float s = 0.f;
  for (int c = tid; c < Dq; c += 256) s += xr[c];
  red[tid] = s; __syncthreads();
  for (int o = 128; o > 0; o >>= 1) { if (tid < o) red[tid] += red[tid + o]; __syncthreads(); }
  const float mean = red[0] / (float)Dq; __syncthreads();
  float v = 0.f;
  for (int c = tid; c < Dq; c += 256) { float d = xr[c] - mean; v += d * d; }
  red[tid] = v; __syncthreads();
  for (int o = 128; o > 0; o >>= 1) { if (tid < o) red[tid] += red[tid + o]; __syncthreads(); }
  const float rstd = rsqrtf(red[0] / (float)Dq + 1e-5f);
  const int l = row % Lq;
  for (int c = tid; c < Dq; c += 256) {
    float val = (xr[c] - mean) * rstd * w[c] + b[c];
    if (extra) val += extra[(long long)l * Dq + c];
    out[(long long)row * Dq + c] = f2bf(val);
  }
}

// ---------------------------------------------------------------------------
// qkv [B,L,3,H,HD] bf16 -> q[B,H,L,HD], k[B,H,L,HD], vT[B,H,HD,L]
// ---------------------------------------------------------------------------
__global__ void reshape_qkv(const unsigned short* __restrict__ qkv,
                            unsigned short* __restrict__ q,
                            unsigned short* __restrict__ k,
                            unsigned short* __restrict__ vT)
{
  long long i = (long long)blockIdx.x * blockDim.x + threadIdx.x;
  const long long total = (long long)Mrows * 3 * Dq;
  if (i >= total) return;
  const int c  = (int)(i % (3 * Dq));
  const long long bl = i / (3 * Dq);
  const int l  = (int)(bl % Lq);
  const int b  = (int)(bl / Lq);
  const int three = c / Dq;
  const int h  = (c % Dq) / HDq;
  const int hd = c % HDq;
  const unsigned short val = qkv[i];
  const long long bh = (long long)(b * Hq + h);
  if (three == 0)      q [(bh * Lq + l) * HDq + hd] = val;
  else if (three == 1) k [(bh * Lq + l) * HDq + hd] = val;
  else                 vT[(bh * HDq + hd) * Lq + l] = val;
}

// ---------------------------------------------------------------------------
// In-place row softmax over width L (attention probabilities, f32 in d_out)
// ---------------------------------------------------------------------------
__global__ void __launch_bounds__(256) softmax_rows(float* __restrict__ p)
{
  __shared__ float red[256];
  float* r = p + (long long)blockIdx.x * Lq;
  const int tid = threadIdx.x;
  float mx = -3.4e38f;
  for (int c = tid; c < Lq; c += 256) mx = fmaxf(mx, r[c]);
  red[tid] = mx; __syncthreads();
  for (int o = 128; o > 0; o >>= 1) { if (tid < o) red[tid] = fmaxf(red[tid], red[tid + o]); __syncthreads(); }
  mx = red[0]; __syncthreads();
  float s = 0.f;
  for (int c = tid; c < Lq; c += 256) { float e = __expf(r[c] - mx); r[c] = e; s += e; }
  red[tid] = s; __syncthreads();
  for (int o = 128; o > 0; o >>= 1) { if (tid < o) red[tid] += red[tid + o]; __syncthreads(); }
  const float inv = 1.f / red[0];
  for (int c = tid; c < Lq; c += 256) r[c] *= inv;
}

// ---------------------------------------------------------------------------
// Gated residual combine: out = sig(g)*xv + (1-sig(g))*res
// ---------------------------------------------------------------------------
__global__ void gate_combine(const float* __restrict__ g,
                             const float* __restrict__ xv,
                             const float* __restrict__ res,
                             float* __restrict__ out, long long total)
{
  long long i = (long long)blockIdx.x * blockDim.x + threadIdx.x;
  if (i >= total) return;
  const float gs = sigmoidf(g[i]);
  out[i] = gs * xv[i] + (1.f - gs) * res[i];
}

// ---------------------------------------------------------------------------
// Adaptive decomposition. freq_filter is a scalar c -> irfft(rfft(x)*c)==c*x.
// ---------------------------------------------------------------------------
__device__ __forceinline__ float dwconv(const float* col, int l,
                                        const float* w, int k, float cf)
{
  float a = 0.f; const int half = k >> 1;
  for (int t = 0; t < k; ++t) {
    const int ll = l + t - half;
    if (ll >= 0 && ll < Lq) a += w[t] * col[(long long)ll * Dq];
  }
  return a * cf;
}

__global__ void decomp_kernel(const float* __restrict__ xg,
    const float* __restrict__ fw, const float* __restrict__ ff,
    const float* __restrict__ cw0, const float* __restrict__ cw1,
    const float* __restrict__ cw2, const float* __restrict__ cw3,
    const float* __restrict__ cb0, const float* __restrict__ cb1,
    const float* __restrict__ cb2, const float* __restrict__ cb3,
    float* __restrict__ seas_f, unsigned short* __restrict__ seas_b)
{
  const long long total = (long long)Mrows * Dq;
  long long idx = (long long)blockIdx.x * blockDim.x + threadIdx.x;
  if (idx >= total) return;
  const int d = (int)(idx % Dq);
  const int l = (int)((idx / Dq) % Lq);
  const float e0 = __expf(fw[0]), e1 = __expf(fw[1]), e2 = __expf(fw[2]), e3 = __expf(fw[3]);
  const float inv = 1.f / (e0 + e1 + e2 + e3);
  const float w0 = e0 * inv, w1 = e1 * inv, w2 = e2 * inv, w3 = e3 * inv;
  const float cf = ff[0];
  const float* col = xg + (idx - (long long)l * Dq);  // (b, 0, d)
  float t = w0 * cb0[d] + w1 * cb1[d] + w2 * cb2[d] + w3 * cb3[d];
  t += w0 * dwconv(col, l, cw0 + d * 3,  3,  cf);
  t += w1 * dwconv(col, l, cw1 + d * 7,  7,  cf);
  t += w2 * dwconv(col, l, cw2 + d * 15, 15, cf);
  t += w3 * dwconv(col, l, cw3 + d * 31, 31, cf);
  const float s = xg[idx] - t;
  seas_f[idx] = s;
  seas_b[idx] = f2bf(s);
}

// ---------------------------------------------------------------------------
// GLU: act[m,j] = u[m,j] * sigmoid(u[m, DFF + j]); u is [M, 2*DFF] bf16
// ---------------------------------------------------------------------------
__global__ void glu_kernel(const unsigned short* __restrict__ u,
                           unsigned short* __restrict__ act)
{
  const long long total = (long long)Mrows * DFFq;
  long long i = (long long)blockIdx.x * blockDim.x + threadIdx.x;
  if (i >= total) return;
  const long long m = i / DFFq;
  const int j = (int)(i % DFFq);
  const float a = bf2f(u[m * (2 * DFFq) + j]);
  const float g = bf2f(u[m * (2 * DFFq) + DFFq + j]);
  act[i] = f2bf(a * sigmoidf(g));
}

// ---------------------------------------------------------------------------
// Host side
// ---------------------------------------------------------------------------
static inline unsigned cdivu(long long a, int b) { return (unsigned)((a + b - 1) / b); }
static inline char* bump(char*& p, size_t bytes) {
  char* r = p; p += (bytes + 255) & ~(size_t)255; return r;
}

extern "C" void kernel_launch(void* const* d_in, const int* in_sizes, int n_in,
                              void* d_out, int out_size, void* d_ws, size_t ws_size,
                              hipStream_t stream)
{
  (void)in_sizes; (void)n_in; (void)out_size; (void)ws_size;
  const float* x      = (const float*)d_in[0];
  const float* n1w    = (const float*)d_in[1];
  const float* n1b    = (const float*)d_in[2];
  const float* pw1    = (const float*)d_in[3];
  const float* pb1    = (const float*)d_in[4];
  const float* pw2    = (const float*)d_in[5];
  const float* pb2    = (const float*)d_in[6];
  const float* qkv_w  = (const float*)d_in[7];
  const float* qkv_b  = (const float*)d_in[8];
  const float* temp   = (const float*)d_in[9];
  const float* out_w  = (const float*)d_in[10];
  const float* out_b  = (const float*)d_in[11];
  const float* gate_w = (const float*)d_in[12];
  const float* gate_b = (const float*)d_in[13];
  const float* freq_w = (const float*)d_in[14];
  const float* freq_f = (const float*)d_in[15];
  const float* cw0 = (const float*)d_in[16];
  const float* cw1 = (const float*)d_in[17];
  const float* cw2 = (const float*)d_in[18];
  const float* cw3 = (const float*)d_in[19];
  const float* cb0 = (const float*)d_in[20];
  const float* cb1 = (const float*)d_in[21];
  const float* cb2 = (const float*)d_in[22];
  const float* cb3 = (const float*)d_in[23];
  const float* f1w = (const float*)d_in[24];
  const float* f1b = (const float*)d_in[25];
  const float* f2w = (const float*)d_in[26];
  const float* f2b = (const float*)d_in[27];
  const float* n2w = (const float*)d_in[28];
  const float* n2b = (const float*)d_in[29];

  float* y_out = (float*)d_out;
  float* attn_probs = (float*)d_out + (long long)Mrows * Dq;   // [B,H,L,L] f32

  // workspace layout
  char* p = (char*)d_ws;
  unsigned short* wt_qkv  = (unsigned short*)bump(p, (size_t)3 * Dq * Dq * 2);
  unsigned short* wt_out  = (unsigned short*)bump(p, (size_t)Dq * Dq * 2);
  unsigned short* wt_gate = (unsigned short*)bump(p, (size_t)Dq * 2 * Dq * 2);
  unsigned short* wt_f1   = (unsigned short*)bump(p, (size_t)2 * DFFq * Dq * 2);
  unsigned short* wt_f2   = (unsigned short*)bump(p, (size_t)Dq * DFFq * 2);
  float*          posemb  = (float*)bump(p, (size_t)Lq * Dq * 4);
  unsigned short* x_bf    = (unsigned short*)bump(p, (size_t)Mrows * Dq * 2);
  unsigned short* h_bf    = (unsigned short*)bump(p, (size_t)Mrows * Dq * 2);
  unsigned short* qkv_bf  = (unsigned short*)bump(p, (size_t)Mrows * 3 * Dq * 2);
  unsigned short* q_bf    = (unsigned short*)bump(p, (size_t)Mrows * Dq * 2);
  unsigned short* k_bf    = (unsigned short*)bump(p, (size_t)Mrows * Dq * 2);
  unsigned short* vT_bf   = (unsigned short*)bump(p, (size_t)Mrows * Dq * 2);
  unsigned short* ao_bf   = (unsigned short*)bump(p, (size_t)Mrows * Dq * 2);  // attn@V, [B,L,D]
  float*          proj_f  = (float*)bump(p, (size_t)Mrows * Dq * 4);
  unsigned short* proj_b  = (unsigned short*)bump(p, (size_t)Mrows * Dq * 2);
  float*          g1_f    = (float*)bump(p, (size_t)Mrows * Dq * 4);
  float*          xg_f    = (float*)bump(p, (size_t)Mrows * Dq * 4);
  float*          seas_f  = (float*)bump(p, (size_t)Mrows * Dq * 4);
  unsigned short* seas_b  = (unsigned short*)bump(p, (size_t)Mrows * Dq * 2);
  unsigned short* xn2_b   = (unsigned short*)bump(p, (size_t)Mrows * Dq * 2);
  unsigned short* u_bf    = (unsigned short*)bump(p, (size_t)Mrows * 2 * DFFq * 2);
  unsigned short* act_bf  = (unsigned short*)bump(p, (size_t)Mrows * DFFq * 2);
  float*          ffo_f   = (float*)bump(p, (size_t)Mrows * Dq * 4);
  unsigned short* ffo_b   = (unsigned short*)bump(p, (size_t)Mrows * Dq * 2);
  float*          g2_f    = (float*)bump(p, (size_t)Mrows * Dq * 4);

  // 1. weight transpose+convert (f32 [K,N] -> bf16 [N,K])
  transpose_to_bf16<<<cdivu((long long)Dq * 3 * Dq, 256), 256, 0, stream>>>(qkv_w,  wt_qkv,  Dq, 3 * Dq);
  transpose_to_bf16<<<cdivu((long long)Dq * Dq,     256), 256, 0, stream>>>(out_w,  wt_out,  Dq, Dq);
  transpose_to_bf16<<<cdivu((long long)2 * Dq * Dq, 256), 256, 0, stream>>>(gate_w, wt_gate, 2 * Dq, Dq);
  transpose_to_bf16<<<cdivu((long long)Dq * 2 * DFFq, 256), 256, 0, stream>>>(f1w,  wt_f1,   Dq, 2 * DFFq);
  transpose_to_bf16<<<cdivu((long long)DFFq * Dq,   256), 256, 0, stream>>>(f2w,   wt_f2,   DFFq, Dq);

  // 2. positional embedding + residual bf16 copy + LN1 (+pos)
  pos_emb_kernel<<<Lq, 128, 0, stream>>>(pw1, pb1, pw2, pb2, posemb);
  cvt_bf16<<<cdivu((long long)Mrows * Dq, 256), 256, 0, stream>>>(x, x_bf, (long long)Mrows * Dq);
  ln_rows<<<Mrows, 256, 0, stream>>>(x, n1w, n1b, posemb, h_bf);

  // 3. QKV projection: [8192,512] @ [512,1536] -> bf16
  {
    unsigned tiles = (Mrows / 32) * (3 * Dq / 32);
    wmma_gemm<false, false><<<dim3((tiles + 7) / 8, 1, 1), 256, 0, stream>>>(
        h_bf, h_bf, nullptr, wt_qkv, qkv_b, nullptr, qkv_bf,
        Mrows, 3 * Dq, Dq, Dq, Dq, Dq, 3 * Dq,
        0, 0, 0, 0, 1, nullptr, 1.0f);
  }
  reshape_qkv<<<cdivu((long long)Mrows * 3 * Dq, 256), 256, 0, stream>>>(qkv_bf, q_bf, k_bf, vT_bf);

  // 4. attention scores: per (b,h): Q[1024,64] @ K^T -> d_out attn region (f32)
  {
    unsigned tiles = (Lq / 32) * (Lq / 32);
    wmma_gemm<false, false><<<dim3((tiles + 7) / 8, 1, Bq * Hq), 256, 0, stream>>>(
        q_bf, q_bf, nullptr, k_bf, nullptr, attn_probs, nullptr,
        Lq, Lq, HDq, HDq, HDq, HDq, Lq,
        (long long)Lq * HDq, (long long)Lq * HDq,
        (long long)Lq * Lq, 0, 1, temp, 0.125f /* 1/sqrt(64) */);
  }

  // 5. softmax in place (rows = B*H*L)
  softmax_rows<<<Bq * Hq * Lq, 256, 0, stream>>>(attn_probs);

  // 6. P @ V: per (b,h): attn[1024,1024](f32) @ V -> attn_out bf16 in [B,L,H*HD]
  {
    unsigned tiles = (Lq / 32) * (HDq / 32);
    wmma_gemm<true, false><<<dim3((tiles + 7) / 8, 1, Bq * Hq), 256, 0, stream>>>(
        nullptr, nullptr, attn_probs, vT_bf, nullptr, nullptr, ao_bf,
        Lq, HDq, Lq, Lq, Lq, Lq, Dq,
        (long long)Lq * Lq, (long long)HDq * Lq,
        (long long)Lq * Dq, HDq, Hq, nullptr, 1.0f);
  }

  // 7. output projection (f32 + bf16 outputs)
  {
    unsigned tiles = (Mrows / 32) * (Dq / 32);
    wmma_gemm<false, false><<<dim3((tiles + 7) / 8, 1, 1), 256, 0, stream>>>(
        ao_bf, ao_bf, nullptr, wt_out, out_b, proj_f, proj_b,
        Mrows, Dq, Dq, Dq, Dq, Dq, Dq,
        0, 0, 0, 0, 1, nullptr, 1.0f);
  }

  // 8. gate 1: [proj, x] @ gate_w -> g1 ; xg = sig(g1)*proj + (1-sig)*x
  {
    unsigned tiles = (Mrows / 32) * (Dq / 32);
    wmma_gemm<false, true><<<dim3((tiles + 7) / 8, 1, 1), 256, 0, stream>>>(
        proj_b, x_bf, nullptr, wt_gate, gate_b, g1_f, nullptr,
        Mrows, Dq, 2 * Dq, Dq, Dq, 2 * Dq, Dq,
        0, 0, 0, 0, 1, nullptr, 1.0f);
  }
  gate_combine<<<cdivu((long long)Mrows * Dq, 256), 256, 0, stream>>>(
      g1_f, proj_f, x, xg_f, (long long)Mrows * Dq);

  // 9. decomposition (scalar freq filter + weighted depthwise convs)
  decomp_kernel<<<cdivu((long long)Mrows * Dq, 256), 256, 0, stream>>>(
      xg_f, freq_w, freq_f, cw0, cw1, cw2, cw3, cb0, cb1, cb2, cb3,
      seas_f, seas_b);

  // 10. LN2 -> bf16
  ln_rows<<<Mrows, 256, 0, stream>>>(seas_f, n2w, n2b, nullptr, xn2_b);

  // 11. FFN up (GLU): [8192,512] @ [512,4096] -> u bf16
  {
    unsigned tiles = (Mrows / 32) * (2 * DFFq / 32);
    wmma_gemm<false, false><<<dim3((tiles + 7) / 8, 1, 1), 256, 0, stream>>>(
        xn2_b, xn2_b, nullptr, wt_f1, f1b, nullptr, u_bf,
        Mrows, 2 * DFFq, Dq, Dq, Dq, Dq, 2 * DFFq,
        0, 0, 0, 0, 1, nullptr, 1.0f);
  }
  glu_kernel<<<cdivu((long long)Mrows * DFFq, 256), 256, 0, stream>>>(u_bf, act_bf);

  // 12. FFN down: [8192,2048] @ [2048,512]
  {
    unsigned tiles = (Mrows / 32) * (Dq / 32);
    wmma_gemm<false, false><<<dim3((tiles + 7) / 8, 1, 1), 256, 0, stream>>>(
        act_bf, act_bf, nullptr, wt_f2, f2b, ffo_f, ffo_b,
        Mrows, Dq, DFFq, DFFq, DFFq, DFFq, Dq,
        0, 0, 0, 0, 1, nullptr, 1.0f);
  }

  // 13. gate 2: [ffn_out, seasonal] @ gate_w -> g2 ; y = blend -> d_out
  {
    unsigned tiles = (Mrows / 32) * (Dq / 32);
    wmma_gemm<false, true><<<dim3((tiles + 7) / 8, 1, 1), 256, 0, stream>>>(
        ffo_b, seas_b, nullptr, wt_gate, gate_b, g2_f, nullptr,
        Mrows, Dq, 2 * Dq, Dq, Dq, 2 * Dq, Dq,
        0, 0, 0, 0, 1, nullptr, 1.0f);
  }
  gate_combine<<<cdivu((long long)Mrows * Dq, 256), 256, 0, stream>>>(
      g2_f, ffo_f, seas_f, y_out, (long long)Mrows * Dq);
}
